// ContextualizationLayer_13176959664435
// MI455X (gfx1250) — compile-verified, matrix operand order checked by reading, not run on hardware
//
#include <hip/hip_runtime.h>

// ---------- types ----------
typedef __attribute__((ext_vector_type(16))) __bf16 bf16x16;
typedef __attribute__((ext_vector_type(8)))  __bf16 bf16x8;
typedef __attribute__((ext_vector_type(8)))  float  f32x8;

#define Bc 2
#define Lc 2048
#define Dc 768
#define Kc 12
#define HDc 64

// ---------- helpers ----------
__device__ __forceinline__ bf16x16 load_fragA(const __bf16* base, int ld, int lane) {
    // 16-bit A-matrix 16x32 layout (ISA 7.12.2):
    // row = lane&15 ; lanes 0-15 hold K=[0..7]+[16..23], lanes 16-31 hold K=[8..15]+[24..31]
    int r  = lane & 15;
    int c0 = (lane >> 4) * 8;
    const __bf16* p = base + r * ld + c0;
    union { bf16x16 v; bf16x8 h[2]; } u;
    u.h[0] = *(const bf16x8*)(p);
    u.h[1] = *(const bf16x8*)(p + 16);
    return u.v;
}

__device__ __forceinline__ f32x8 wmma_bf16(bf16x16 a, bf16x16 b, f32x8 c) {
    return __builtin_amdgcn_wmma_f32_16x16x32_bf16(
        /*neg_a=*/false, a, /*neg_b=*/false, b,
        /*c_mod=*/(short)0, c, /*reuse_a=*/false, /*reuse_b=*/false);
}

// ---------- kernel 1: f32 -> bf16 elementwise ----------
__global__ void cvt_f32_bf16(const float* __restrict__ in, __bf16* __restrict__ out, int n) {
    int i = blockIdx.x * blockDim.x + threadIdx.x;
    if (i < n) out[i] = (__bf16)in[i];
}

// ---------- kernel 2: W (c,n) f32 -> WT (n,c) bf16 ----------
__global__ void cvt_wT(const float* __restrict__ W, __bf16* __restrict__ WT) {
    int idx = blockIdx.x * blockDim.x + threadIdx.x;
    if (idx >= Dc * Dc) return;
    int n = idx / Dc, c = idx % Dc;
    WT[idx] = (__bf16)W[c * Dc + n];
}

// ---------- kernel 3: C_x (b,j,d,k) f32 -> Ct (b,k,d,j) bf16, tiled via LDS ----------
__global__ void transpose_cx(const float* __restrict__ Cx, __bf16* __restrict__ Ct) {
    __shared__ __bf16 tile[Kc * 32 * 32];   // [h][d][j]  24 KB
    int d0 = blockIdx.x * 32;
    int j0 = blockIdx.y * 32;
    int b  = blockIdx.z;
    const float* src = Cx + (size_t)(b * Lc + j0) * (Dc * Kc);
    // coalesced read: contiguous (d,k) span per j row
    for (int t = threadIdx.x; t < 32 * 32 * Kc; t += blockDim.x) {
        int j = t / (32 * Kc);
        int r = t % (32 * Kc);
        int d = r / Kc;
        int h = r % Kc;
        float v = src[(size_t)j * (Dc * Kc) + (d0 + d) * Kc + h];
        tile[(h * 32 + d) * 32 + j] = (__bf16)v;
    }
    __syncthreads();
    for (int t = threadIdx.x; t < 32 * 32 * Kc; t += blockDim.x) {
        int h = t >> 10;
        int r = t & 1023;
        int d = r >> 5;
        int j = r & 31;
        Ct[((size_t)(b * Kc + h) * Dc + d0 + d) * Lc + j0 + j] = tile[(h * 32 + d) * 32 + j];
    }
}

// ---------- kernel 4: projection q/k = x @ W  (one wave -> 16x64 tile) ----------
__global__ void proj_qk(const __bf16* __restrict__ xb,
                        const __bf16* __restrict__ wqT, const __bf16* __restrict__ wkT,
                        __bf16* __restrict__ qb, __bf16* __restrict__ kb) {
    int lane = threadIdx.x;
    int i0 = blockIdx.x * 16;     // row tile in L
    int h  = blockIdx.y;          // head == 64-wide n-chunk
    int b  = blockIdx.z >> 1;
    int sel = blockIdx.z & 1;     // 0 = q, 1 = k
    const __bf16* X = xb + (size_t)(b * Lc + i0) * Dc;
    const __bf16* W = sel ? wkT : wqT;   // (n, c) row-major
    f32x8 acc[4] = {};
    for (int c0 = 0; c0 < Dc; c0 += 32) {
        bf16x16 a = load_fragA(X + c0, Dc, lane);
        #pragma unroll
        for (int dt = 0; dt < 4; ++dt) {
            bf16x16 bf = load_fragA(W + (size_t)(h * HDc + dt * 16) * Dc + c0, Dc, lane);
            acc[dt] = wmma_bf16(a, bf, acc[dt]);
        }
    }
    __bf16* O = (sel ? kb : qb) + ((size_t)(b * Kc + h) * Lc + i0) * HDc;
    int n = lane & 15, hi = lane >> 4;
    #pragma unroll
    for (int dt = 0; dt < 4; ++dt)
        #pragma unroll
        for (int r = 0; r < 8; ++r)
            O[(r + hi * 8) * HDc + dt * 16 + n] = (__bf16)acc[dt][r];
}

// ---------- kernel 5: scores + softmax -> alpha (bf16) ----------
// block = 128 threads (4 waves) per (b, head, 16-row tile); dyn LDS = 16*2048 f32 + stats
__global__ void scores_softmax(const __bf16* __restrict__ qb, const __bf16* __restrict__ kb,
                               __bf16* __restrict__ alpha) {
    extern __shared__ char smemraw[];
    float* S       = (float*)smemraw;           // [16][2048]
    float* partial = S + 16 * Lc;               // [16][8]
    float* rstat   = partial + 128;             // [16] (max, then 1/sum)
    int t = threadIdx.x, lane = t & 31, wave = t >> 5;
    int i0 = blockIdx.x * 16, h = blockIdx.y, b = blockIdx.z;
    const __bf16* Q  = qb + ((size_t)(b * Kc + h) * Lc + i0) * HDc;
    const __bf16* Kp = kb + ((size_t)(b * Kc + h) * Lc) * HDc;
    bf16x16 a0 = load_fragA(Q,      HDc, lane);
    bf16x16 a1 = load_fragA(Q + 32, HDc, lane);
    int n = lane & 15, hi = lane >> 4;
    for (int jt = wave; jt < Lc / 16; jt += 4) {
        f32x8 acc = {};
        bf16x16 b0 = load_fragA(Kp + (size_t)jt * 16 * HDc,      HDc, lane);
        bf16x16 b1 = load_fragA(Kp + (size_t)jt * 16 * HDc + 32, HDc, lane);
        acc = wmma_bf16(a0, b0, acc);
        acc = wmma_bf16(a1, b1, acc);
        #pragma unroll
        for (int r = 0; r < 8; ++r)
            S[(r + hi * 8) * Lc + jt * 16 + n] = acc[r] * 0.125f;  // 1/sqrt(64)
    }
    __syncthreads();
    // row max (8 threads per row, 256 cols each)
    int row = t >> 3, seg = t & 7;
    float* Sr = S + row * Lc + seg * 256;
    float mx = -__builtin_inff();
    for (int i = 0; i < 256; ++i) mx = fmaxf(mx, Sr[i]);
    partial[row * 8 + seg] = mx;
    __syncthreads();
    if (t < 16) {
        float m = -__builtin_inff();
        for (int s = 0; s < 8; ++s) m = fmaxf(m, partial[t * 8 + s]);
        rstat[t] = m;
    }
    __syncthreads();
    float rm = rstat[row], sum = 0.f;
    for (int i = 0; i < 256; ++i) { float e = __expf(Sr[i] - rm); Sr[i] = e; sum += e; }
    __syncthreads();   // all reads of rstat (max) done before overwrite
    partial[row * 8 + seg] = sum;
    __syncthreads();
    if (t < 16) {
        float s = 0.f;
        for (int q = 0; q < 8; ++q) s += partial[t * 8 + q];
        rstat[t] = 1.0f / s;
    }
    __syncthreads();
    // normalized write, coalesced over columns
    __bf16* A = alpha + ((size_t)(b * Kc + h) * Lc + i0) * Lc;
    for (int idx = t; idx < 16 * Lc; idx += 128) {
        int r2 = idx >> 11, c = idx & (Lc - 1);
        A[(size_t)r2 * Lc + c] = (__bf16)(S[idx] * rstat[r2]);
    }
}

// ---------- kernel 6: o[b,i,d] = sum_{h,j} alpha[b,h,i,j] * Ct[b,h,d,j] ----------
// one wave -> 16x64 output tile; Ct (75MB bf16) is L2-resident across i-tiles
__global__ void out_gemm(const __bf16* __restrict__ alpha, const __bf16* __restrict__ Ct,
                         float* __restrict__ out) {
    int lane = threadIdx.x;
    int i0 = blockIdx.x * 16;
    int d0 = blockIdx.y * 64;
    int b  = blockIdx.z;
    f32x8 acc[4] = {};
    for (int h = 0; h < Kc; ++h) {
        const __bf16* A  = alpha + ((size_t)(b * Kc + h) * Lc + i0) * Lc;
        const __bf16* Bt = Ct    + ((size_t)(b * Kc + h) * Dc + d0) * Lc;
        for (int jt = 0; jt < Lc / 32; ++jt) {
            bf16x16 a = load_fragA(A + jt * 32, Lc, lane);
            #pragma unroll
            for (int dt = 0; dt < 4; ++dt) {
                bf16x16 bf = load_fragA(Bt + (size_t)(dt * 16) * Lc + jt * 32, Lc, lane);
                acc[dt] = wmma_bf16(a, bf, acc[dt]);
            }
        }
    }
    float* O = out + ((size_t)(b * Lc) + i0) * Dc + d0;
    int n = lane & 15, hi = lane >> 4;
    #pragma unroll
    for (int dt = 0; dt < 4; ++dt)
        #pragma unroll
        for (int r = 0; r < 8; ++r)
            O[(size_t)(r + hi * 8) * Dc + dt * 16 + n] = acc[dt][r];
}

// ---------- launch ----------
extern "C" void kernel_launch(void* const* d_in, const int* in_sizes, int n_in,
                              void* d_out, int out_size, void* d_ws, size_t ws_size,
                              hipStream_t stream) {
    (void)in_sizes; (void)n_in; (void)out_size;
    const float* x  = (const float*)d_in[0];   // (B,L,D)
    const float* Cx = (const float*)d_in[1];   // (B,L,D,K)
    const float* Wq = (const float*)d_in[2];   // (D, K*HD)
    const float* Wk = (const float*)d_in[3];
    float* out = (float*)d_out;                // (B,L,D)

    auto align = [](size_t v) { return (v + 255) & ~(size_t)255; };
    char* ws = (char*)d_ws;
    size_t off = 0;
    const size_t N_X  = (size_t)Bc * Lc * Dc;             // 3.1M
    const size_t N_W  = (size_t)Dc * Dc;                  // 0.59M
    const size_t N_QK = (size_t)Bc * Kc * Lc * HDc;       // 3.1M
    const size_t N_CT = (size_t)Bc * Kc * Dc * Lc;        // 37.7M
    const size_t N_AL = (size_t)Bc * Kc * Lc * Lc;        // 100.7M

    __bf16* xb   = (__bf16*)(ws + off); off = align(off + N_X  * 2);
    __bf16* wqT  = (__bf16*)(ws + off); off = align(off + N_W  * 2);
    __bf16* wkT  = (__bf16*)(ws + off); off = align(off + N_W  * 2);
    __bf16* qb   = (__bf16*)(ws + off); off = align(off + N_QK * 2);
    __bf16* kb   = (__bf16*)(ws + off); off = align(off + N_QK * 2);
    __bf16* Ct   = (__bf16*)(ws + off); off = align(off + N_CT * 2);
    __bf16* alph = (__bf16*)(ws + off); off = align(off + N_AL * 2);
    if (off > ws_size) { /* workspace too small; nothing safe to do */ }

    cvt_f32_bf16<<<(int)((N_X + 255) / 256), 256, 0, stream>>>(x, xb, (int)N_X);
    cvt_wT<<<(int)((N_W + 255) / 256), 256, 0, stream>>>(Wq, wqT);
    cvt_wT<<<(int)((N_W + 255) / 256), 256, 0, stream>>>(Wk, wkT);
    transpose_cx<<<dim3(Dc / 32, Lc / 32, Bc), 256, 0, stream>>>(Cx, Ct);
    proj_qk<<<dim3(Lc / 16, Kc, Bc * 2), 32, 0, stream>>>(xb, wqT, wkT, qb, kb);
    size_t smem = (16 * Lc + 128 + 16) * sizeof(float);
    scores_softmax<<<dim3(Lc / 16, Kc, Bc), 128, smem, stream>>>(qb, kb, alph);
    out_gemm<<<dim3(Lc / 16, Dc / 64, Bc), 32, 0, stream>>>(alph, Ct, out);
}
